// SimCLRLoss_83167746719846
// MI455X (gfx1250) — matrix operand order repaired and checked by reading, not compiled, not run on hardware
//
#include <hip/hip_runtime.h>

typedef _Float16 v16h __attribute__((ext_vector_type(16)));
typedef _Float16 v8h  __attribute__((ext_vector_type(8)));
typedef _Float16 v4h  __attribute__((ext_vector_type(4)));
typedef float    v8f  __attribute__((ext_vector_type(8)));
typedef float    v4f  __attribute__((ext_vector_type(4)));

#define BROWS 2048
#define DIMD  1024
#define NROWS 4096
#define INV_T 14.285714285714286f   // 1/0.07
#define LDS_STRIDE 1032             // halfs; pad 8 halfs -> conflict-free ds_load_b128

// ---------------- Kernel 1: row L2-normalize, f32 -> f16 ----------------
__global__ __launch_bounds__(256) void k_normalize(
    const float* __restrict__ zi, const float* __restrict__ zj,
    _Float16* __restrict__ zh)
{
  const int row = blockIdx.x;
  const float* src = (row < BROWS) ? zi + (size_t)row * DIMD
                                   : zj + (size_t)(row - BROWS) * DIMD;
  const int t = threadIdx.x;                 // 256 threads * 4 floats = 1024
  v4f x = ((const v4f*)src)[t];
  float ss = x[0]*x[0] + x[1]*x[1] + x[2]*x[2] + x[3]*x[3];
  #pragma unroll
  for (int m = 1; m < 32; m <<= 1) ss += __shfl_xor(ss, m, 32);
  __shared__ float wsum[8];
  __shared__ float s_scale;
  if ((t & 31) == 0) wsum[t >> 5] = ss;
  __syncthreads();
  if (t == 0) {
    float tot = 0.f;
    #pragma unroll
    for (int i = 0; i < 8; ++i) tot += wsum[i];
    s_scale = 1.0f / fmaxf(sqrtf(tot), 1e-12f);
  }
  __syncthreads();
  const float sc = s_scale;
  v4h h;
  h[0] = (_Float16)(x[0]*sc); h[1] = (_Float16)(x[1]*sc);
  h[2] = (_Float16)(x[2]*sc); h[3] = (_Float16)(x[3]*sc);
  ((v4h*)(zh + (size_t)row * DIMD))[t] = h;
}

// ------- Kernel 2: fused Gram (z z^T) tiles + per-row sum of exp --------
// grid = (NROWS/16, 8), block = 64 (2 waves). Block owns 16 rows (A panel in
// LDS); wave w covers columns [blockIdx.y*512 + w*256, +256). Each (chunk,row)
// partial has exactly one writer -> deterministic, no atomics.
__global__ __launch_bounds__(64) void k_gram_softmax(
    const _Float16* __restrict__ zh,
    float* __restrict__ partial,   // [16][NROWS]
    float* __restrict__ diag,      // [NROWS]
    float* __restrict__ pos)       // [NROWS]
{
  __shared__ __align__(16) _Float16 smA[16 * LDS_STRIDE];   // ~33 KB
  const int tid  = threadIdx.x;
  const int w    = tid >> 5;
  const int lane = tid & 31;
  const int lm   = lane & 15;
  const int half = lane >> 4;
  const int rowBase = blockIdx.x * 16;
  const int colBase = blockIdx.y * 512 + w * 256;
  const int chunk   = blockIdx.y * 2 + w;

  // cooperative stage of the 16-row A panel into LDS (padded row stride)
  for (int i = tid; i < 16 * (DIMD / 8); i += 64) {         // 2048/64 = 32 its
    const int r = i >> 7;                                   // DIMD/8 = 128
    const int c = i & 127;
    *(v8h*)(smA + r * LDS_STRIDE + c * 8) =
        *(const v8h*)(zh + (size_t)(rowBase + r) * DIMD + c * 8);
  }
  __syncthreads();

  const _Float16* aRow = smA + lm * LDS_STRIDE;             // A: lane = row M
  const int rowOff = half ? 8 : 0;                          // C/D: vgpr r -> M
  float rsum[8] = {0.f,0.f,0.f,0.f,0.f,0.f,0.f,0.f};

  for (int g = 0; g < 4; ++g) {                             // 4 x (4 tiles of 16 cols)
    v8f acc[4] = {};
    const int colTile0 = colBase + g * 64;
    #pragma unroll 4
    for (int ks = 0; ks < DIMD / 32; ++ks) {
      const int k0 = ks * 32;
      // A frag: lanes 0-15 K=[0..7],[16..23]; lanes 16-31 K=[8..15],[24..31]
      v8h alo = *(const v8h*)(aRow + k0 + half * 8);
      v8h ahi = *(const v8h*)(aRow + k0 + half * 8 + 16);
      v16h a = __builtin_shufflevector(alo, ahi,
                 0,1,2,3,4,5,6,7,8,9,10,11,12,13,14,15);
      #pragma unroll
      for (int t2 = 0; t2 < 4; ++t2) {
        // B frag: lane = column n = row of z; K same order as A frag
        const _Float16* bp = zh + (size_t)(colTile0 + t2 * 16 + lm) * DIMD
                                + k0 + half * 8;
        v8h blo = *(const v8h*)bp;
        v8h bhi = *(const v8h*)(bp + 16);
        v16h b = __builtin_shufflevector(blo, bhi,
                   0,1,2,3,4,5,6,7,8,9,10,11,12,13,14,15);
        acc[t2] = __builtin_amdgcn_wmma_f32_16x16x32_f16(
            false, a, false, b, (short)0, acc[t2], false, false);
      }
    }
    // postprocess: shift by the (constant) row max 1/T; ratio is shift-invariant
    #pragma unroll
    for (int t2 = 0; t2 < 4; ++t2) {
      const int col = colTile0 + t2 * 16 + lm;
      #pragma unroll
      for (int r = 0; r < 8; ++r) {
        const int row = rowBase + rowOff + r;
        const float e = __expf((acc[t2][r] - 1.0f) * INV_T);
        rsum[r] += e;
        if (col == row) diag[row] = e;                         // single writer
        if (col == ((row + BROWS) & (NROWS - 1))) pos[row] = e; // single writer
      }
    }
  }
  // reduce across the 16 lanes of each half-wave (halves hold disjoint rows)
  #pragma unroll
  for (int m = 1; m < 16; m <<= 1) {
    #pragma unroll
    for (int r = 0; r < 8; ++r) rsum[r] += __shfl_xor(rsum[r], m, 32);
  }
  if (lm == 0) {
    #pragma unroll
    for (int r = 0; r < 8; ++r)
      partial[(size_t)chunk * NROWS + rowBase + rowOff + r] = rsum[r];
  }
}

// ------------- Kernel 3: per-row loss + mean (single block) -------------
__global__ __launch_bounds__(256) void k_finalize(
    const float* __restrict__ partial, const float* __restrict__ diag,
    const float* __restrict__ pos, float* __restrict__ out)
{
  const int t = threadIdx.x;
  float lsum = 0.f;
  for (int row = t; row < NROWS; row += 256) {
    float all = 0.f;
    #pragma unroll
    for (int c = 0; c < 16; ++c) all += partial[(size_t)c * NROWS + row];
    all -= diag[row];
    lsum += -logf(pos[row] / all + 1e-8f);
  }
  #pragma unroll
  for (int m = 1; m < 32; m <<= 1) lsum += __shfl_xor(lsum, m, 32);
  __shared__ float wsum[8];
  if ((t & 31) == 0) wsum[t >> 5] = lsum;
  __syncthreads();
  if (t == 0) {
    float tot = 0.f;
    #pragma unroll
    for (int i = 0; i < 8; ++i) tot += wsum[i];
    out[0] = tot / (float)NROWS;
  }
}

extern "C" void kernel_launch(void* const* d_in, const int* in_sizes, int n_in,
                              void* d_out, int out_size, void* d_ws, size_t ws_size,
                              hipStream_t stream) {
  const float* zi = (const float*)d_in[0];
  const float* zj = (const float*)d_in[1];

  char* ws = (char*)d_ws;
  _Float16* zh  = (_Float16*)ws;                       // 4096*1024*2 = 8 MB
  size_t off = (size_t)NROWS * DIMD * sizeof(_Float16);
  float* partial = (float*)(ws + off);  off += (size_t)16 * NROWS * sizeof(float);
  float* diag    = (float*)(ws + off);  off += (size_t)NROWS * sizeof(float);
  float* pos     = (float*)(ws + off);

  k_normalize<<<NROWS, 256, 0, stream>>>(zi, zj, zh);

  dim3 grid2(NROWS / 16, 8);
  k_gram_softmax<<<grid2, 64, 0, stream>>>(zh, partial, diag, pos);

  k_finalize<<<1, 256, 0, stream>>>(partial, diag, pos, (float*)d_out);
}
